// MyLSTMLayer_2826088481192
// MI455X (gfx1250) — compile-verified, matrix operand order checked by reading, not run on hardware
//
#include <hip/hip_runtime.h>

// ---------------------------------------------------------------------------
// LSTM layer for MI455X (gfx1250): bf16 WMMA (v_wmma_f32_16x16x32_bf16),
// fp32 accumulate, fp32 gate preactivations, async LDS staging in the
// recurrent step. B=128, T=512, D=512, H=512.
// ---------------------------------------------------------------------------

#define BB   128
#define TT   512
#define DD   512
#define HH   512
#define NG   (4 * HH)        // fused gate columns: [i | f | c | o]
#define M1   (BB * TT)       // 65536 rows of x

typedef __bf16  v16bf __attribute__((ext_vector_type(16)));
typedef float   v8f   __attribute__((ext_vector_type(8)));

union FragBf {
    uint4  u[2];   // two 16B chunks per lane
    v16bf  v;
};

__device__ __forceinline__ unsigned short f2bf(float f) {
    unsigned int u = __float_as_uint(f);
    unsigned int r = (u >> 16) & 1u;     // round to nearest even
    u += 0x7fffu + r;
    return (unsigned short)(u >> 16);
}

// ---- async global->LDS (CDNA5 path) with safe fallback --------------------
#if __has_builtin(__builtin_amdgcn_global_load_async_to_lds_b128)
#define HAVE_ASYNC_LDS 1
__device__ __forceinline__ void async_cp16(const void* gsrc, void* ldst) {
    typedef int v4i __attribute__((vector_size(16)));
    typedef __attribute__((address_space(1))) v4i* gp_t;
    typedef __attribute__((address_space(3))) v4i* lp_t;
    // generic LDS pointer's low 32 bits are the LDS byte offset (aperture rule)
    gp_t g = (gp_t)(uintptr_t)gsrc;
    lp_t l = (lp_t)(unsigned int)(uintptr_t)ldst;
    __builtin_amdgcn_global_load_async_to_lds_b128(g, l, 0, 0);
}
__device__ __forceinline__ void wait_async0() {
#if __has_builtin(__builtin_amdgcn_s_wait_asynccnt)
    __builtin_amdgcn_s_wait_asynccnt(0);
#else
    asm volatile("s_wait_asynccnt 0x0" ::: "memory");
#endif
}
#else
#define HAVE_ASYNC_LDS 0
#endif

// --------------------------- prep kernels ----------------------------------

// fp32 x -> bf16 xb, 4 elements / thread
__global__ __launch_bounds__(256) void k_cvt_x(const float* __restrict__ x,
                                               unsigned short* __restrict__ xb) {
    size_t i = ((size_t)blockIdx.x * 256 + threadIdx.x) * 4;
    float4 f = *(const float4*)(x + i);
    ushort4 o;
    o.x = f2bf(f.x); o.y = f2bf(f.y); o.z = f2bf(f.z); o.w = f2bf(f.w);
    *(ushort4*)(xb + i) = o;
}

// Pack 4 [512 x 512] fp32 k-major matrices into one [2048 x 512] bf16
// n-major (transposed) matrix: dst[ng*512 + k] = A_g[k*512 + nn].
__global__ __launch_bounds__(256) void k_pack(unsigned short* __restrict__ dst,
                                              const float* __restrict__ A0,
                                              const float* __restrict__ A1,
                                              const float* __restrict__ A2,
                                              const float* __restrict__ A3) {
    int idx = blockIdx.x * 256 + threadIdx.x;   // 0 .. 2048*512-1
    int ng  = idx >> 9;
    int k   = idx & 511;
    int g   = ng >> 9;
    int nn  = ng & 511;
    const float* A = (g == 0) ? A0 : (g == 1) ? A1 : (g == 2) ? A2 : A3;
    dst[idx] = f2bf(A[(size_t)k * HH + nn]);
}

__global__ __launch_bounds__(256) void k_init(unsigned short* __restrict__ hA,
                                              float* __restrict__ c) {
    int i = blockIdx.x * 256 + threadIdx.x;     // 65536 threads
    hA[i] = 0;
    c[i]  = 0.f;
}

// --------------------------- phase 1: input projections --------------------
// G[t][g][b][nn] = x[b,t,:] @ W_g + bias_g   (fp32 output)
// Grid: (NG/128, M1/128) = (16, 512); block 256 = 8 waves.
// WG tile 128x128, wave tile 32x64 (2x4 WMMA tiles).

#define LDP 40   // LDS row stride in bf16 elems (80B = 20 banks: conflict-free)

__global__ __launch_bounds__(256) void k_proj(
        const unsigned short* __restrict__ xb,   // [M1][512] bf16
        const unsigned short* __restrict__ wt,   // [2048][512] bf16
        const float* __restrict__ bi, const float* __restrict__ bf_,
        const float* __restrict__ bc, const float* __restrict__ bo,
        float* __restrict__ G) {
    __shared__ unsigned short As[128 * LDP];
    __shared__ unsigned short Bs[128 * LDP];

    const int tid  = threadIdx.x;
    const int lane = tid & 31;
    const int w    = tid >> 5;
    const int wm   = w >> 1;          // 0..3
    const int wn   = w & 1;           // 0..1
    const int m0   = blockIdx.y * 128;
    const int n0   = blockIdx.x * 128;
    const int half = lane >> 4;
    const int l16  = lane & 15;

    v8f acc[2][4];
    const v8f vzero = {0.f, 0.f, 0.f, 0.f, 0.f, 0.f, 0.f, 0.f};
#pragma unroll
    for (int i = 0; i < 2; ++i)
#pragma unroll
        for (int j = 0; j < 4; ++j) acc[i][j] = vzero;

    for (int k0 = 0; k0 < DD; k0 += 32) {
        // cooperative stage: 128 rows x 64B = 512 x 16B chunks per tile
#pragma unroll
        for (int j = 0; j < 2; ++j) {
            int chunk = tid + j * 256;
            int r   = chunk >> 2;
            int c16 = chunk & 3;
            *(uint4*)(As + r * LDP + c16 * 8) =
                *(const uint4*)(xb + (size_t)(m0 + r) * DD + k0 + c16 * 8);
            *(uint4*)(Bs + r * LDP + c16 * 8) =
                *(const uint4*)(wt + (size_t)(n0 + r) * DD + k0 + c16 * 8);
        }
        __syncthreads();

        FragBf a[2], b[4];
#pragma unroll
        for (int i = 0; i < 2; ++i) {
            const uint4* p = (const uint4*)(As + (wm * 32 + i * 16 + l16) * LDP);
            a[i].u[0] = p[half];      // K = 8h .. 8h+7
            a[i].u[1] = p[2 + half];  // K = 16+8h .. 16+8h+7
        }
#pragma unroll
        for (int j = 0; j < 4; ++j) {
            const uint4* p = (const uint4*)(Bs + (wn * 64 + j * 16 + l16) * LDP);
            b[j].u[0] = p[half];
            b[j].u[1] = p[2 + half];
        }
#pragma unroll
        for (int i = 0; i < 2; ++i)
#pragma unroll
            for (int j = 0; j < 4; ++j)
                acc[i][j] = __builtin_amdgcn_wmma_f32_16x16x32_bf16(
                    false, a[i].v, false, b[j].v, (short)0, acc[i][j],
                    false, false);
        __syncthreads();
    }

    // epilogue: add bias, scatter to G[t][g][b][nn]
#pragma unroll
    for (int i = 0; i < 2; ++i) {
#pragma unroll
        for (int j = 0; j < 4; ++j) {
            int n  = n0 + wn * 64 + j * 16 + l16;
            int g  = n >> 9;
            int nn = n & 511;
            const float* bp = (g == 0) ? bi : (g == 1) ? bf_ : (g == 2) ? bc : bo;
            float bias = bp[nn];
#pragma unroll
            for (int v = 0; v < 8; ++v) {
                int m  = m0 + wm * 32 + i * 16 + half * 8 + v;
                int bb = m >> 9;          // batch (x is [B,T,D], m = b*T + t)
                int t  = m & 511;
                G[(((size_t)t * 4 + g) * BB + bb) * HH + nn] = acc[i][j][v] + bias;
            }
        }
    }
}

// --------------------------- phase 2: recurrent step -----------------------
// One launch per timestep. Grid: (HH/16, BB/64) = (32, 2); block 256.
// WG owns rows mb0..mb0+63 and cols n0..n0+15 for ALL 4 gates.
// wave w: gate g = w>>1, M-half (w&1): two 16x16 accum tiles.
// K is staged in 2 panels of 256 (33KB LDS) -> only 4 barriers per step
// (vs 32 with 32-wide panels), which matters on the serial critical path.

#define KP   256              // K panel
#define HSP  (KP + 8)         // Hs row stride in elems: 528B -> bank stride 4

__global__ __launch_bounds__(256) void k_step(
        const unsigned short* __restrict__ h_in,   // [128][512] bf16
        unsigned short* __restrict__ h_out,        // [128][512] bf16
        float* __restrict__ c,                     // [128][512] fp32
        const float* __restrict__ G,               // [T][4][128][512] fp32
        const unsigned short* __restrict__ ut,     // [2048][512] bf16
        int t,
        float* __restrict__ h_final) {             // d_out fp32
    __shared__ unsigned short Hs[64 * HSP];        // ~33 KB
    __shared__ float Pre[4][64][16];               // 16 KB

    const int tid  = threadIdx.x;
    const int lane = tid & 31;
    const int w    = tid >> 5;
    const int g    = w >> 1;
    const int mh   = (w & 1) * 32;
    const int half = lane >> 4;
    const int l16  = lane & 15;
    const int n0   = blockIdx.x * 16;
    const int mb0  = blockIdx.y * 64;

    // warm L2 for this step's G slab portion (global_prefetch_b8)
    __builtin_prefetch(&G[(((size_t)t * 4) * BB + mb0) * HH + n0], 0, 3);

    v8f acc[2];
    const v8f vzero = {0.f, 0.f, 0.f, 0.f, 0.f, 0.f, 0.f, 0.f};
    acc[0] = vzero; acc[1] = vzero;

    for (int s = 0; s < HH / KP; ++s) {
        // stage h panel: 64 rows x 512B = 2048 x 16B chunks, 8 per thread
#pragma unroll
        for (int j = 0; j < 8; ++j) {
            int chunk = tid + j * 256;
            int r   = chunk >> 5;          // 32 chunks per row
            int c16 = chunk & 31;
            const void* src = h_in + (size_t)(mb0 + r) * HH + s * KP + c16 * 8;
            void*       dst = Hs + r * HSP + c16 * 8;
#if HAVE_ASYNC_LDS
            async_cp16(src, dst);
#else
            *(uint4*)dst = *(const uint4*)src;
#endif
        }
#if HAVE_ASYNC_LDS
        wait_async0();
#endif
        __syncthreads();

#pragma unroll
        for (int kk = 0; kk < KP / 32; ++kk) {
            FragBf a[2], b;
            const uint4* bp = (const uint4*)(ut +
                (size_t)(g * HH + n0 + l16) * HH + s * KP + kk * 32);
            b.u[0] = bp[half];
            b.u[1] = bp[2 + half];
#pragma unroll
            for (int i = 0; i < 2; ++i) {
                const uint4* p = (const uint4*)(Hs + (mh + i * 16 + l16) * HSP);
                a[i].u[0] = p[kk * 4 + half];
                a[i].u[1] = p[kk * 4 + 2 + half];
            }
#pragma unroll
            for (int i = 0; i < 2; ++i)
                acc[i] = __builtin_amdgcn_wmma_f32_16x16x32_bf16(
                    false, a[i].v, false, b.v, (short)0, acc[i], false, false);
        }
        __syncthreads();
    }

    // write h@U preacts to LDS, grouped by gate
#pragma unroll
    for (int i = 0; i < 2; ++i)
#pragma unroll
        for (int v = 0; v < 8; ++v)
            Pre[g][mh + i * 16 + half * 8 + v][l16] = acc[i][v];
    __syncthreads();

    // elementwise: 64x16 = 1024 elements, 4 per thread (coalesced over nn)
#pragma unroll
    for (int j = 0; j < 4; ++j) {
        int e  = tid + j * 256;
        int bl = e >> 4;
        int nl = e & 15;
        int bb = mb0 + bl;
        int nn = n0 + nl;
        size_t gb = (((size_t)t * 4) * BB + bb) * HH + nn;
        const size_t gstride = (size_t)BB * HH;
        float pi = Pre[0][bl][nl] + G[gb];
        float pf = Pre[1][bl][nl] + G[gb + gstride];
        float pc = Pre[2][bl][nl] + G[gb + 2 * gstride];
        float po = Pre[3][bl][nl] + G[gb + 3 * gstride];
        float ig = 1.f / (1.f + __expf(-pi));
        float fg = 1.f / (1.f + __expf(-pf));
        float og = 1.f / (1.f + __expf(-po));
        float ch = tanhf(pc);
        size_t ci = (size_t)bb * HH + nn;
        float cn = fg * c[ci] + ig * ch;
        c[ci] = cn;
        float hn = og * tanhf(cn);
        h_out[ci] = f2bf(hn);
        if (t == TT - 1) h_final[ci] = hn;   // layer output = final h
    }
}

// --------------------------- host launcher ---------------------------------

extern "C" void kernel_launch(void* const* d_in, const int* in_sizes, int n_in,
                              void* d_out, int out_size, void* d_ws, size_t ws_size,
                              hipStream_t stream) {
    const float* x  = (const float*)d_in[0];
    const float* Wi = (const float*)d_in[1];
    const float* Ui = (const float*)d_in[2];
    const float* bi = (const float*)d_in[3];
    const float* Wf = (const float*)d_in[4];
    const float* Uf = (const float*)d_in[5];
    const float* bf = (const float*)d_in[6];
    const float* Wc = (const float*)d_in[7];
    const float* Uc = (const float*)d_in[8];
    const float* bc = (const float*)d_in[9];
    const float* Wo = (const float*)d_in[10];
    const float* Uo = (const float*)d_in[11];
    const float* bo = (const float*)d_in[12];

    char* ws = (char*)d_ws;
    size_t off = 0;
    unsigned short* xb = (unsigned short*)(ws + off); off += (size_t)M1 * DD * 2;       // 64 MB
    unsigned short* Wt = (unsigned short*)(ws + off); off += (size_t)NG * DD * 2;       // 2 MB
    unsigned short* Ut = (unsigned short*)(ws + off); off += (size_t)NG * HH * 2;       // 2 MB
    float*          G  = (float*)(ws + off);          off += (size_t)TT * 4 * BB * HH * 4; // 512 MB
    unsigned short* hA = (unsigned short*)(ws + off); off += (size_t)BB * HH * 2;
    unsigned short* hB = (unsigned short*)(ws + off); off += (size_t)BB * HH * 2;
    float*          cS = (float*)(ws + off);          off += (size_t)BB * HH * 4;

    k_cvt_x<<<(M1 * DD) / (256 * 4), 256, 0, stream>>>(x, xb);
    k_pack<<<(NG * DD) / 256, 256, 0, stream>>>(Wt, Wi, Wf, Wc, Wo);
    k_pack<<<(NG * HH) / 256, 256, 0, stream>>>(Ut, Ui, Uf, Uc, Uo);
    k_init<<<(BB * HH) / 256, 256, 0, stream>>>(hA, cS);

    k_proj<<<dim3(NG / 128, M1 / 128), 256, 0, stream>>>(xb, Wt, bi, bf, bc, bo, G);

    for (int t = 0; t < TT; ++t) {
        unsigned short* hin  = (t & 1) ? hB : hA;
        unsigned short* hout = (t & 1) ? hA : hB;
        k_step<<<dim3(HH / 16, BB / 64), 256, 0, stream>>>(
            hin, hout, cS, G, Ut, t, (float*)d_out);
    }
}
